// GaussianActor_39960375722686
// MI455X (gfx1250) — matrix operand order repaired and checked by reading, not compiled, not run on hardware
//
#include <hip/hip_runtime.h>

// ---------------------------------------------------------------------------
// GaussianActor forward on gfx1250: bf16 WMMA GEMMs with TDM (tensor_load_to_lds)
// double-buffered LDS staging for the A operand.
// ---------------------------------------------------------------------------

#define BDIM   32768
#define OBS    512
#define HID    1024
#define ACT    64
#define NHEAD  8

#define USE_TDM 1

typedef __attribute__((ext_vector_type(16))) __bf16 v16bf;
typedef __attribute__((ext_vector_type(8)))  __bf16 v8bf;
typedef __attribute__((ext_vector_type(8)))  float  v8f;
typedef __attribute__((ext_vector_type(4)))  unsigned int u32x4;
typedef __attribute__((ext_vector_type(8)))  unsigned int u32x8;

// ------------------------------ fp32 -> bf16 -------------------------------
__global__ void cvt_f32_bf16(const float* __restrict__ src,
                             __bf16* __restrict__ dst, int n) {
    int i = blockIdx.x * blockDim.x + threadIdx.x;
    if (i < n) dst[i] = (__bf16)src[i];
}

// -------------------- pack weights into WMMA B-operand ---------------------
// B fragment (K=32 x N=16 bf16): lane l holds column n = l%16;
// lanes 0-15 -> K=0..7,16..23, lanes 16-31 -> K=8..15,24..31; 16 bf16 per lane.
__global__ void pack_w_kernel(const float* __restrict__ W,
                              __bf16* __restrict__ dst,
                              int K, int N, int head_mode) {
    int p = blockIdx.x * blockDim.x + threadIdx.x;
    if (p >= K * N) return;
    int j    = p & 15;
    int lane = (p >> 4) & 31;
    int frag = p >> 9;
    int ktiles = K >> 5;
    int kt = frag % ktiles;
    int nt = frag / ktiles;
    int k = kt * 32 + ((j >> 3) << 4) + ((lane >> 4) << 3) + (j & 7);
    int n = nt * 16 + (lane & 15);
    float v;
    if (head_mode) {
        int s = n >> 7, o = n & 127;           // src = Wh[s, h=k, o]
        v = W[((size_t)s * HID + k) * 128 + o];
    } else {
        v = W[(size_t)k * N + n];
    }
    dst[p] = (__bf16)v;
}

// ------------------------------- WMMA GEMM ---------------------------------
// C[M,1024] = A[M,K] @ Wp + bias ; EPI: 0 = bias, 1 = bias+leaky, 2 = raw
#define BM 128
#define BN 128
#define BK 32
#define LDA 40   // LDS row stride in elements: 64B row + 16B pad (TDM pad feature)

template <int EPI>
__global__ __launch_bounds__(256) void gemm_bf16(
    const __bf16* __restrict__ A, const __bf16* __restrict__ Bp,
    const float* __restrict__ bias, __bf16* __restrict__ C, int K) {
    __shared__ alignas(16) __bf16 At[2][BM * LDA];

    const int tid  = threadIdx.x;
    const int lane = tid & 31;
    const int wave = tid >> 5;
    const int wm = wave & 3;       // 4 waves along M (4*32 = 128)
    const int wn = wave >> 2;      // 2 waves along N (2*64 = 128)
    const int m0 = blockIdx.x * BM;
    const int n0 = blockIdx.y * BN;
    const int ktiles = K >> 5;
    const int nt_base = (n0 >> 4) + wn * 4;

    v8f acc[2][4] = {};

#if USE_TDM
    // -------- Tensor Data Mover: DMA 128x32 bf16 tile into LDS (wave 0) ----
    const unsigned lds0 = (unsigned)(size_t)&At[0][0];
    const unsigned lds1 = (unsigned)(size_t)&At[1][0];
    auto tdm_issue = [&](int kt, int buf) {
        unsigned long long ga =
            (unsigned long long)(const void*)A +
            ((size_t)m0 * (size_t)K + (size_t)kt * 32) * 2ull;
        u32x4 g0;
        g0[0] = 1u;                                   // count=1, load descriptor
        g0[1] = buf ? lds1 : lds0;                    // lds_addr (bytes)
        g0[2] = (unsigned)ga;                         // global_addr[31:0]
        g0[3] = ((unsigned)(ga >> 32) & 0x01FFFFFFu)  // global_addr[56:32]
                | (2u << 30);                         // type = 2 (image)
        u32x8 g1;
        g1[0] = (1u << 16)      // data_size = 2 bytes
              | (1u << 20)      // pad_enable
              | (3u << 22)      // pad_interval: every 16 dwords (64B row)
              | (3u << 25);     // pad_amount: 4 dwords (16B) -> 80B row stride
        unsigned td0 = (unsigned)K;       // tensor_dim0
        unsigned td1 = 1u << 20;          // tensor_dim1 (no OOB)
        g1[1] = (td0 & 0xFFFFu) << 16;                    // dim0[15:0]
        g1[2] = (td0 >> 16) | ((td1 & 0xFFFFu) << 16);    // dim0[31:16], dim1[15:0]
        g1[3] = (td1 >> 16) | (32u << 16);                // dim1[31:16], tile_dim0=32
        g1[4] = 128u;                                     // tile_dim1=128, tile_dim2=0
        g1[5] = (unsigned)K;                              // dim0_stride[31:0] (elems)
        g1[6] = 0u;                                       // stride[47:32], dim1_stride lo
        g1[7] = 0u;
        u32x4 gz;
        gz[0] = 0u; gz[1] = 0u; gz[2] = 0u; gz[3] = 0u;
        asm volatile("tensor_load_to_lds %0, %1, %2, %3"
                     :: "s"(g0), "s"(g1), "s"(gz), "s"(gz)
                     : "memory");
    };

    if (wave == 0) tdm_issue(0, 0);
    for (int kt = 0; kt < ktiles; ++kt) {
        const int buf = kt & 1;
        if (wave == 0) __builtin_amdgcn_s_wait_tensorcnt(0);
        __syncthreads();  // releases only after wave0 saw the DMA complete
        if (wave == 0 && kt + 1 < ktiles) tdm_issue(kt + 1, buf ^ 1);
        // A fragments (16x32): lane holds row m=lane%16, kbase=(lane/16)*8
        v16bf af[2];
#pragma unroll
        for (int i = 0; i < 2; ++i) {
            int mrow = wm * 32 + i * 16 + (lane & 15);
            int kb   = (lane >> 4) << 3;
            v8bf lo = *(const v8bf*)(&At[buf][mrow * LDA + kb]);
            v8bf hi = *(const v8bf*)(&At[buf][mrow * LDA + 16 + kb]);
            af[i] = __builtin_shufflevector(lo, hi, 0, 1, 2, 3, 4, 5, 6, 7, 8,
                                            9, 10, 11, 12, 13, 14, 15);
        }
#pragma unroll
        for (int j = 0; j < 4; ++j) {
            size_t fragIdx = (size_t)(nt_base + j) * ktiles + kt;
            v16bf bfr = *(const v16bf*)(Bp + fragIdx * 512 + lane * 16);
#pragma unroll
            for (int i = 0; i < 2; ++i) {
                acc[i][j] = __builtin_amdgcn_wmma_f32_16x16x32_bf16(
                    false, af[i], false, bfr, (short)0, acc[i][j], false, false);
            }
        }
        // no trailing barrier: next tile lands in the other buffer; reuse of
        // this buffer is fenced by the next iteration's tensorcnt+barrier.
    }
#else
    for (int kt = 0; kt < ktiles; ++kt) {
        const int k0 = kt << 5;
#pragma unroll
        for (int r = 0; r < 2; ++r) {
            int c   = tid + r * 256;
            int row = c >> 2;
            int ch  = c & 3;
            const v8bf* src =
                (const v8bf*)(A + (size_t)(m0 + row) * K + k0 + ch * 8);
            *(v8bf*)(&At[0][row * LDA + ch * 8]) = *src;
        }
        __syncthreads();
        v16bf af[2];
#pragma unroll
        for (int i = 0; i < 2; ++i) {
            int mrow = wm * 32 + i * 16 + (lane & 15);
            int kb   = (lane >> 4) << 3;
            v8bf lo = *(const v8bf*)(&At[0][mrow * LDA + kb]);
            v8bf hi = *(const v8bf*)(&At[0][mrow * LDA + 16 + kb]);
            af[i] = __builtin_shufflevector(lo, hi, 0, 1, 2, 3, 4, 5, 6, 7, 8,
                                            9, 10, 11, 12, 13, 14, 15);
        }
#pragma unroll
        for (int j = 0; j < 4; ++j) {
            size_t fragIdx = (size_t)(nt_base + j) * ktiles + kt;
            v16bf bfr = *(const v16bf*)(Bp + fragIdx * 512 + lane * 16);
#pragma unroll
            for (int i = 0; i < 2; ++i) {
                acc[i][j] = __builtin_amdgcn_wmma_f32_16x16x32_bf16(
                    false, af[i], false, bfr, (short)0, acc[i][j], false, false);
            }
        }
        __syncthreads();
    }
#endif

    // epilogue: C/D layout -> lane holds n=lane%16, rows r + (lane/16)*8
#pragma unroll
    for (int j = 0; j < 4; ++j) {
        int ng = n0 + wn * 64 + j * 16 + (lane & 15);
        float bv = (EPI == 2) ? 0.0f : bias[ng];
#pragma unroll
        for (int i = 0; i < 2; ++i) {
            int mg0 = m0 + wm * 32 + i * 16 + ((lane >> 4) << 3);
#pragma unroll
            for (int r = 0; r < 8; ++r) {
                float v = acc[i][j][r] + bv;
                if (EPI == 1) v = v >= 0.0f ? v : 0.01f * v;
                C[(size_t)(mg0 + r) * HID + ng] = (__bf16)v;
            }
        }
    }
}

// -------------------------- LayerNorm + LeakyReLU --------------------------
__global__ __launch_bounds__(256) void ln_lrelu_kernel(
    const __bf16* __restrict__ H, const float* __restrict__ lw,
    const float* __restrict__ lb, __bf16* __restrict__ Y) {
    __shared__ float s1[256];
    __shared__ float s2[256];
    const int r = blockIdx.x;
    const __bf16* row = H + (size_t)r * HID;
    float x[4], sum = 0.f, sq = 0.f;
#pragma unroll
    for (int t = 0; t < 4; ++t) {
        x[t] = (float)row[threadIdx.x + t * 256];
        sum += x[t];
        sq  += x[t] * x[t];
    }
    s1[threadIdx.x] = sum;
    s2[threadIdx.x] = sq;
    __syncthreads();
    for (int s = 128; s > 0; s >>= 1) {
        if (threadIdx.x < s) {
            s1[threadIdx.x] += s1[threadIdx.x + s];
            s2[threadIdx.x] += s2[threadIdx.x + s];
        }
        __syncthreads();
    }
    float mean = s1[0] * (1.0f / HID);
    float var  = s2[0] * (1.0f / HID) - mean * mean;
    float inv  = rsqrtf(var + 1e-5f);
#pragma unroll
    for (int t = 0; t < 4; ++t) {
        int c = threadIdx.x + t * 256;
        float y = (x[t] - mean) * inv * lw[c] + lb[c];
        y = y >= 0.0f ? y : 0.01f * y;
        Y[(size_t)r * HID + c] = (__bf16)y;
    }
}

// ------------------- per-sample head select + bias + clip ------------------
__global__ __launch_bounds__(128) void head_gather_kernel(
    const __bf16* __restrict__ Hout, const int* __restrict__ stage,
    const float* __restrict__ bh, float* __restrict__ out) {
    const int r = blockIdx.x;
    const int o = threadIdx.x;  // 0..127
    const int s = stage[r];
    float v = (float)Hout[(size_t)r * HID + s * 128 + o] + bh[s * 128 + o];
    if (o < 64) {
        out[(size_t)r * 64 + o] = v;  // mean
    } else {
        v = fminf(fmaxf(v, -20.0f), 2.0f);
        out[(size_t)BDIM * 64 + (size_t)r * 64 + (o - 64)] = v;  // log_std
    }
}

// ------------------------------- launcher ----------------------------------
extern "C" void kernel_launch(void* const* d_in, const int* in_sizes, int n_in,
                              void* d_out, int out_size, void* d_ws,
                              size_t ws_size, hipStream_t stream) {
    const float* obs  = (const float*)d_in[0];
    const int*   stg  = (const int*)d_in[1];
    const float* W0   = (const float*)d_in[2];
    const float* b0   = (const float*)d_in[3];
    const float* ln_w = (const float*)d_in[4];
    const float* ln_b = (const float*)d_in[5];
    const float* W1   = (const float*)d_in[6];
    const float* b1   = (const float*)d_in[7];
    const float* W2   = (const float*)d_in[8];
    const float* b2   = (const float*)d_in[9];
    const float* W3   = (const float*)d_in[10];
    const float* b3   = (const float*)d_in[11];
    const float* Wh   = (const float*)d_in[12];
    const float* bh   = (const float*)d_in[13];
    float* out = (float*)d_out;

    char* ws = (char*)d_ws;
    const size_t sz_obs = (size_t)BDIM * OBS * 2;
    const size_t sz_w0  = (size_t)OBS * HID * 2;
    const size_t sz_w   = (size_t)HID * HID * 2;
    const size_t sz_act = (size_t)BDIM * HID * 2;
    size_t off = 0;
    __bf16* obs_bf = (__bf16*)(ws + off); off += sz_obs;
    __bf16* w0p    = (__bf16*)(ws + off); off += sz_w0;
    __bf16* w1p    = (__bf16*)(ws + off); off += sz_w;
    __bf16* w2p    = (__bf16*)(ws + off); off += sz_w;
    __bf16* w3p    = (__bf16*)(ws + off); off += sz_w;
    __bf16* whp    = (__bf16*)(ws + off); off += sz_w;
    __bf16* act_a  = (__bf16*)(ws + off); off += sz_act;
    __bf16* act_b  = (__bf16*)(ws + off); off += sz_act;
    (void)ws_size; (void)in_sizes; (void)n_in; (void)out_size;

    {
        int n = BDIM * OBS;
        cvt_f32_bf16<<<(n + 255) / 256, 256, 0, stream>>>(obs, obs_bf, n);
    }
    pack_w_kernel<<<(OBS * HID + 255) / 256, 256, 0, stream>>>(W0, w0p, OBS, HID, 0);
    pack_w_kernel<<<(HID * HID + 255) / 256, 256, 0, stream>>>(W1, w1p, HID, HID, 0);
    pack_w_kernel<<<(HID * HID + 255) / 256, 256, 0, stream>>>(W2, w2p, HID, HID, 0);
    pack_w_kernel<<<(HID * HID + 255) / 256, 256, 0, stream>>>(W3, w3p, HID, HID, 0);
    pack_w_kernel<<<(HID * HID + 255) / 256, 256, 0, stream>>>(Wh, whp, HID, HID, 1);

    dim3 ggrid(BDIM / BM, HID / BN);
    gemm_bf16<0><<<ggrid, 256, 0, stream>>>(obs_bf, w0p, b0, act_a, OBS);   // pre-LN
    ln_lrelu_kernel<<<BDIM, 256, 0, stream>>>(act_a, ln_w, ln_b, act_b);
    gemm_bf16<1><<<ggrid, 256, 0, stream>>>(act_b, w1p, b1, act_a, HID);
    gemm_bf16<1><<<ggrid, 256, 0, stream>>>(act_a, w2p, b2, act_b, HID);
    gemm_bf16<0><<<ggrid, 256, 0, stream>>>(act_b, w3p, b3, act_a, HID);
    gemm_bf16<2><<<ggrid, 256, 0, stream>>>(act_a, whp, nullptr, act_b, HID);
    head_gather_kernel<<<BDIM, 128, 0, stream>>>(act_b, stg, bh, out);
}
